// RowWiseGatedAttention_89661737271787
// MI455X (gfx1250) — matrix operand order repaired and checked by reading, not compiled, not run on hardware
//
#include <hip/hip_runtime.h>
#include <hip/hip_fp16.h>

// ---------------------------------------------------------------------------
// RowWiseGatedAttention for MI455X (gfx1250).
// Identity: softmax(scores).sum(-1) == 1  =>  out = proj(sig(G)*V) + b.
// GEMMs on v_wmma_f32_16x16x32_f16 (wave32), fragment-native LDS tiles so
// every WMMA operand is exactly two ds_load_b128. Tile staging uses
// GLOBAL_LOAD_ASYNC_TO_LDS_B128 (ASYNCcnt) when the toolchain exposes it.
// ---------------------------------------------------------------------------

typedef __attribute__((ext_vector_type(16))) _Float16 v16h;
typedef __attribute__((ext_vector_type(8)))  float    v8f;
typedef int v4i __attribute__((vector_size(16)));     // matches builtin param type

#define CM     256                  // channels
#define SP     32768                // spatial positions = 128*256
#define NELEM  (CM * SP)            // 8388608
#define LN_EPS 1e-5f
#define ASTR   40                   // LDS row stride in halves (80B, 16B-aligned)

#if defined(__has_builtin)
#  if __has_builtin(__builtin_amdgcn_global_load_async_to_lds_b128)
#    define HAVE_ASYNC_LDS 1
#  endif
#endif
#ifndef HAVE_ASYNC_LDS
#  define HAVE_ASYNC_LDS 0
#endif

// 16-byte global -> LDS tile copy (per-lane). Async path writes LDS directly
// (no VGPR round-trip); completion via ASYNCcnt before the workgroup barrier.
static __device__ __forceinline__ void stage16(_Float16* lds_dst, const _Float16* gsrc) {
#if HAVE_ASYNC_LDS
  v4i* g = (v4i*)gsrc;      // drop const / retype in generic space
  v4i* l = (v4i*)lds_dst;
  __builtin_amdgcn_global_load_async_to_lds_b128(
      (__attribute__((address_space(1))) v4i*)g,
      (__attribute__((address_space(3))) v4i*)l,
      /*offset=*/0, /*cpol=*/0);
#else
  *reinterpret_cast<uint4*>(lds_dst) = *reinterpret_cast<const uint4*>(gsrc);
#endif
}

static __device__ __forceinline__ void stage_wait() {
#if HAVE_ASYNC_LDS
#  if __has_builtin(__builtin_amdgcn_s_wait_asynccnt)
  __builtin_amdgcn_s_wait_asynccnt(0);
#  else
  asm volatile("s_wait_asynccnt 0x0" ::: "memory");
#  endif
#endif
}

static __device__ __forceinline__ v8f wmma_f16(v16h a, v16h b, v8f c) {
  return __builtin_amdgcn_wmma_f32_16x16x32_f16(
      /*neg_a=*/false, a, /*neg_b=*/false, b,
      /*c_mod=*/(short)0, c, /*reuse_a=*/false, /*reuse_b=*/false);
}

// A 16x32 f16 (ISA 7.12.2): lane holds row m; VGPR v pair at
// K = (v&4)<<2 | hi<<3 | (v&3)<<1  ->  two contiguous 8-half runs:
//   halves [hi*8 .. hi*8+7] and [16+hi*8 .. 16+hi*8+7]
static __device__ __forceinline__ v16h frag_a(const _Float16* sA, int m, int hi) {
  union { uint4 q[2]; v16h h; } u;
  const char* base = (const char*)sA + m * (ASTR * 2) + hi * 16;
  u.q[0] = *reinterpret_cast<const uint4*>(base);
  u.q[1] = *reinterpret_cast<const uint4*>(base + 32);
  return u.h;
}

// B 32x16 f16: lane holds col n; same K order as frag_a.
// With the tile stored [n][k] in LDS this is halves [hi*8 .. hi*8+7] and [16+hi*8 .. 16+hi*8+7].
static __device__ __forceinline__ v16h frag_b(const _Float16* sB, int n, int hi) {
  union { uint4 q[2]; v16h h; } u;
  const char* base = (const char*)sB + n * (ASTR * 2) + hi * 16;
  u.q[0] = *reinterpret_cast<const uint4*>(base);
  u.q[1] = *reinterpret_cast<const uint4*>(base + 32);
  return u.h;
}

// ---------------------------------------------------------------------------
// 1) Global LayerNorm statistics (mean/var over all C,H,W of msa_rep)
// ---------------------------------------------------------------------------
__global__ __launch_bounds__(256)
void k_reduce_partial(const float* __restrict__ x,
                      float* __restrict__ part_sum, float* __restrict__ part_sq) {
  __shared__ float s_s[256], s_q[256];
  float s = 0.f, q = 0.f;
  for (int i = blockIdx.x * 256 + threadIdx.x; i < NELEM; i += gridDim.x * 256) {
    float v = x[i];
    s += v;
    q += v * v;
  }
  s_s[threadIdx.x] = s;
  s_q[threadIdx.x] = q;
  __syncthreads();
  for (int off = 128; off > 0; off >>= 1) {
    if ((int)threadIdx.x < off) {
      s_s[threadIdx.x] += s_s[threadIdx.x + off];
      s_q[threadIdx.x] += s_q[threadIdx.x + off];
    }
    __syncthreads();
  }
  if (threadIdx.x == 0) {
    part_sum[blockIdx.x] = s_s[0];
    part_sq[blockIdx.x]  = s_q[0];
  }
}

__global__ __launch_bounds__(256)
void k_finalize_stats(const float* __restrict__ ps, const float* __restrict__ pq,
                      float* __restrict__ stats /* [mu, rsigma] */) {
  __shared__ float s_s[256], s_q[256];
  const int t = threadIdx.x;
  s_s[t] = ps[t] + ps[t + 256];
  s_q[t] = pq[t] + pq[t + 256];
  __syncthreads();
  for (int off = 128; off > 0; off >>= 1) {
    if (t < off) { s_s[t] += s_s[t + off]; s_q[t] += s_q[t + off]; }
    __syncthreads();
  }
  if (t == 0) {
    const float inv = 1.0f / (float)NELEM;
    const float mu  = s_s[0] * inv;
    const float var = s_q[0] * inv - mu * mu;
    stats[0] = mu;
    stats[1] = rsqrtf(var + LN_EPS);
  }
}

// ---------------------------------------------------------------------------
// 2) Weight prep: wv/wg plain f32->f16 (keep [o][c] = fragment-native [n][k]);
//    w_rep f16 with (h c)->(c h) column permutation baked in (GEMM2 A side).
//    perm(n) = (n%32)*8 + n/32
// ---------------------------------------------------------------------------
__global__ __launch_bounds__(256)
void k_prep_weights(const float* __restrict__ wv, const float* __restrict__ wg,
                    const float* __restrict__ wrep,
                    _Float16* __restrict__ wv_h, _Float16* __restrict__ wg_h,
                    _Float16* __restrict__ wrep_p) {
  const int idx = blockIdx.x * 256 + threadIdx.x;  // 0..65535
  wv_h[idx] = (_Float16)wv[idx];
  wg_h[idx] = (_Float16)wg[idx];
  const int r = idx >> 8;   // cm
  const int n = idx & 255;  // col
  const int perm = ((n & 31) << 3) | (n >> 5);
  wrep_p[idx] = (_Float16)wrep[(r << 8) + perm];
}

// ---------------------------------------------------------------------------
// 3) Fused LayerNorm + transpose:  m_t[p][c] (f16)  <-  msa [c][p] (f32)
// ---------------------------------------------------------------------------
__global__ __launch_bounds__(256)
void k_ln_transpose(const float* __restrict__ x, const float* __restrict__ w,
                    const float* __restrict__ b, const float* __restrict__ stats,
                    _Float16* __restrict__ m_t) {
  __shared__ float tile[32][33];
  const float mu = stats[0], rs = stats[1];
  const int p0 = blockIdx.x * 32;
  const int c0 = blockIdx.y * 32;
  const int tx = threadIdx.x;   // 0..31 along p (coalesced reads)
  const int ty = threadIdx.y;   // 0..7
#pragma unroll
  for (int r = 0; r < 4; ++r) {
    const int c  = ty * 4 + r;
    const int gi = (c0 + c) * SP + p0 + tx;
    tile[c][tx] = (x[gi] - mu) * rs * w[gi] + b[gi];
  }
  __syncthreads();
#pragma unroll
  for (int r = 0; r < 4; ++r) {
    const int pr = ty * 4 + r;
    m_t[(p0 + pr) * 256 + c0 + tx] = (_Float16)tile[tx][pr];
  }
}

// ---------------------------------------------------------------------------
// 4) GEMM-VG: og[p][n] = sigmoid(m_t @ Wg^T + b_g) * (m_t @ Wv^T)   (f16 out)
//    Block 64(M=p) x 64(N=o), 8 waves of 16x32, K loop over 256.
//    B tiles staged [n][k] (== row-major w_v/w_g) -> frag_b is 2x b128.
// ---------------------------------------------------------------------------
__global__ __launch_bounds__(256)
void k_gemm_vg(const _Float16* __restrict__ m_t, const _Float16* __restrict__ wv_h,
               const _Float16* __restrict__ wg_h, const float* __restrict__ b_g,
               _Float16* __restrict__ og) {
  __shared__ _Float16 smem[3 * 64 * ASTR];   // A | Bv | Bg (reused as 64x72 out tile)
  _Float16* sA  = smem;
  _Float16* sBv = smem + 64 * ASTR;
  _Float16* sBg = sBv + 64 * ASTR;

  const int tid  = threadIdx.x;
  const int p0   = blockIdx.x * 64;
  const int n0   = blockIdx.y * 64;
  const int wid  = tid >> 5, lane = tid & 31;
  const int wm   = wid & 3,  wn   = wid >> 2;
  const int hi   = (lane >> 4) & 1, l15 = lane & 15;

  const int trow = tid >> 2, tcol = (tid & 3) * 8;   // 64 rows x 32 halves stages

  v8f av0 = {}, av1 = {}, ag0 = {}, ag1 = {};

  for (int k0 = 0; k0 < 256; k0 += 32) {
    if (k0 < 224)
      __builtin_prefetch(m_t + (p0 + trow) * 256 + (k0 + 32) + tcol, 0, 3);
    stage16(sA + trow * ASTR + tcol, m_t + (p0 + trow) * 256 + k0 + tcol);
    stage16(sBv + trow * ASTR + tcol, wv_h + (n0 + trow) * 256 + k0 + tcol);
    stage16(sBg + trow * ASTR + tcol, wg_h + (n0 + trow) * 256 + k0 + tcol);
    stage_wait();
    __syncthreads();

    const v16h a  = frag_a(sA, wm * 16 + l15, hi);
    const v16h b0 = frag_b(sBv, wn * 32 + l15, hi);
    const v16h b1 = frag_b(sBv, wn * 32 + 16 + l15, hi);
    const v16h g0 = frag_b(sBg, wn * 32 + l15, hi);
    const v16h g1 = frag_b(sBg, wn * 32 + 16 + l15, hi);
    av0 = wmma_f16(a, b0, av0);
    av1 = wmma_f16(a, b1, av1);
    ag0 = wmma_f16(a, g0, ag0);
    ag1 = wmma_f16(a, g1, ag1);
    __syncthreads();
  }

  // Epilogue: gate, stage f16 tile [m][n] (stride 72 => 16B-aligned rows),
  // then write og[p][n] as coalesced b128.
  _Float16* sT = smem;   // 64*72 = 4608 halves, fits
#pragma unroll
  for (int sub = 0; sub < 2; ++sub) {
    const v8f vv = sub ? av1 : av0;
    const v8f gg = sub ? ag1 : ag0;
    const int n_loc = wn * 32 + sub * 16 + l15;
    const float bg = b_g[n0 + n_loc];
#pragma unroll
    for (int r = 0; r < 8; ++r) {
      const int m_loc = wm * 16 + r + hi * 8;
      const float gate = 1.0f / (1.0f + __expf(-(gg[r] + bg)));
      sT[m_loc * 72 + n_loc] = (_Float16)(gate * vv[r]);
    }
  }
  __syncthreads();
#pragma unroll
  for (int c = 0; c < 2; ++c) {
    const int idx = c * 256 + tid;     // 512 chunks of 8 halves
    const int mm = idx >> 3;
    const int ch = (idx & 7) * 8;
    *reinterpret_cast<uint4*>(og + (p0 + mm) * 256 + n0 + ch) =
        *reinterpret_cast<const uint4*>(sT + mm * 72 + ch);
  }
}

// ---------------------------------------------------------------------------
// 5) GEMM-OUT: out[cm][p] = wrep_p[cm][:] @ og[p][:]^T + b_rep[cm]
//    M = cm(256), K = och(256), N = p(32768). og[p][n] is fragment-native
//    [N][K] for the B side. f32 accumulate, coalesced direct stores.
// ---------------------------------------------------------------------------
__global__ __launch_bounds__(256)
void k_gemm_out(const _Float16* __restrict__ wrep_p, const _Float16* __restrict__ og,
                const float* __restrict__ b_rep, float* __restrict__ out) {
  __shared__ _Float16 smem[2 * 64 * ASTR];
  _Float16* sA = smem;
  _Float16* sB = smem + 64 * ASTR;

  const int tid  = threadIdx.x;
  const int m0   = blockIdx.x * 64;   // cm
  const int q0   = blockIdx.y * 64;   // p
  const int wid  = tid >> 5, lane = tid & 31;
  const int wm   = wid & 3,  wn   = wid >> 2;
  const int hi   = (lane >> 4) & 1, l15 = lane & 15;

  const int trow = tid >> 2, tcol = (tid & 3) * 8;

  v8f acc0 = {}, acc1 = {};

  for (int k0 = 0; k0 < 256; k0 += 32) {
    if (k0 < 224)
      __builtin_prefetch(og + (q0 + trow) * 256 + (k0 + 32) + tcol, 0, 3);
    stage16(sA + trow * ASTR + tcol, wrep_p + (m0 + trow) * 256 + k0 + tcol);
    stage16(sB + trow * ASTR + tcol, og + (q0 + trow) * 256 + k0 + tcol);
    stage_wait();
    __syncthreads();

    const v16h a  = frag_a(sA, wm * 16 + l15, hi);
    const v16h b0 = frag_b(sB, wn * 32 + l15, hi);
    const v16h b1 = frag_b(sB, wn * 32 + 16 + l15, hi);
    acc0 = wmma_f16(a, b0, acc0);
    acc1 = wmma_f16(a, b1, acc1);
    __syncthreads();
  }

#pragma unroll
  for (int sub = 0; sub < 2; ++sub) {
    const v8f ac = sub ? acc1 : acc0;
    const int n_loc = wn * 32 + sub * 16 + l15;   // p offset (coalesced across lanes)
#pragma unroll
    for (int r = 0; r < 8; ++r) {
      const int m_loc = wm * 16 + r + hi * 8;     // cm offset
      out[(m0 + m_loc) * SP + q0 + n_loc] = ac[r] + b_rep[m0 + m_loc];
    }
  }
}

// ---------------------------------------------------------------------------
// Launch
// ---------------------------------------------------------------------------
extern "C" void kernel_launch(void* const* d_in, const int* in_sizes, int n_in,
                              void* d_out, int out_size, void* d_ws, size_t ws_size,
                              hipStream_t stream) {
  (void)in_sizes; (void)n_in; (void)out_size; (void)ws_size;

  const float* msa   = (const float*)d_in[0];
  const float* ln1w  = (const float*)d_in[2];
  const float* ln1b  = (const float*)d_in[3];
  const float* w_v   = (const float*)d_in[8];
  const float* w_g   = (const float*)d_in[10];
  const float* b_g   = (const float*)d_in[11];
  const float* w_rep = (const float*)d_in[12];
  const float* b_rep = (const float*)d_in[13];
  float* out = (float*)d_out;

  // Workspace carve (all 16B aligned). Total ~32.5 MB.
  char* ws = (char*)d_ws;
  float*    part_sum = (float*)(ws + 0);                 // 512 f
  float*    part_sq  = (float*)(ws + 2048);              // 512 f
  float*    stats    = (float*)(ws + 4096);              // 2 f
  _Float16* wv_h     = (_Float16*)(ws + 8192);           // 256*256 f16
  _Float16* wg_h     = (_Float16*)(ws + 8192 + 131072);
  _Float16* wrep_p   = (_Float16*)(ws + 8192 + 2 * 131072);
  _Float16* m_t      = (_Float16*)(ws + (1u << 19));               // [p][c] f16 (16 MB)
  _Float16* og       = (_Float16*)(ws + (1u << 19) + 16777216u);   // [p][n] f16 (16 MB)

  k_reduce_partial<<<512, 256, 0, stream>>>(msa, part_sum, part_sq);
  k_finalize_stats<<<1, 256, 0, stream>>>(part_sum, part_sq, stats);
  k_prep_weights<<<256, 256, 0, stream>>>(w_v, w_g, w_rep, wv_h, wg_h, wrep_p);
  k_ln_transpose<<<dim3(SP / 32, CM / 32), dim3(32, 8), 0, stream>>>(msa, ln1w, ln1b, stats, m_t);
  k_gemm_vg<<<dim3(SP / 64, 256 / 64), 256, 0, stream>>>(m_t, wv_h, wg_h, b_g, og);
  k_gemm_out<<<dim3(256 / 64, SP / 64), 256, 0, stream>>>(wrep_p, og, b_rep, out);
}
